// RWKVTimeMixX070_90228672954789
// MI455X (gfx1250) — compile-verified, matrix operand order checked by reading, not run on hardware
//
#include <hip/hip_runtime.h>
#include <hip/hip_bf16.h>
#include <math.h>
#include <stdint.h>

// ---------------------------------------------------------------------------
// Problem constants (RWKV-7 time mix)
// ---------------------------------------------------------------------------
constexpr int Bsz   = 2;
constexpr int S_LEN = 4096;
constexpr int C_DIM = 2048;
constexpr int Hh    = 32;
constexpr int Nh    = 64;
constexpr int Mrows = Bsz * S_LEN;      // 8192
constexpr int DWd   = 128;
constexpr int DAd   = 128;
constexpr int DVd   = 64;
constexpr int DGd   = 224;

#define AS1 __attribute__((address_space(1)))
#define AS3 __attribute__((address_space(3)))

// CDNA5 async global->LDS path (guarded: fall back to ld+ds_store if absent)
#if defined(__HIP_DEVICE_COMPILE__) && defined(__has_builtin)
#if __has_builtin(__builtin_amdgcn_global_load_async_to_lds_b128) && \
    __has_builtin(__builtin_amdgcn_s_wait_asynccnt)
#define ASYNC_OK 1
#endif
#endif
#ifndef ASYNC_OK
#define ASYNC_OK 0
#endif

// ---------------------------------------------------------------------------
// WMMA helpers (gfx1250: v_wmma_f32_16x16x32_bf16, wave32)
// ---------------------------------------------------------------------------
typedef __attribute__((ext_vector_type(16))) __bf16 v16bf;
typedef __attribute__((ext_vector_type(8)))  __bf16 v8bf;
typedef __attribute__((ext_vector_type(8)))  float  v8f;
typedef __attribute__((ext_vector_type(4)))  float  v4f;
typedef __attribute__((ext_vector_type(4)))  int    v4i;

__device__ inline v8f wmma_bf16(v16bf a, v16bf b, v8f c) {
  // 8 args: (neg_a, A, neg_b, B, c_mod, C, reuse_a, reuse_b)
  return __builtin_amdgcn_wmma_f32_16x16x32_bf16(false, a, false, b, (short)0, c,
                                                 false, false);
}

// 16-bit A/B fragment from an LDS tile with row stride 32 (bf16 elements).
// Lane L: row = row0 + (L&15); khalf = (L>>4)*8; holds K[kh..kh+7] and
// K[16+kh..16+kh+7]  (per CDNA5 ISA 16-bit 16x32 layout).
__device__ inline v16bf frag_ld(const __bf16* tile, int row0, int lane) {
  const int m  = row0 + (lane & 15);
  const int kh = (lane >> 4) << 3;
  union { v16bf v; v8bf h[2]; } f;
  f.h[0] = *(const v8bf*)(tile + m * 32 + kh);
  f.h[1] = *(const v8bf*)(tile + m * 32 + 16 + kh);
  return f.v;
}

// ---------------------------------------------------------------------------
// Epilogues
// ---------------------------------------------------------------------------
enum EpKind {
  EP_BF16_NONE    = 0,   // outB = s
  EP_BF16_TANH    = 1,   // outB = tanh(s)
  EP_BF16_SIG     = 2,   // outB = sigmoid(s)
  EP_BF16_SIGBIAS = 3,   // outB = sigmoid(bias[n] + s)
  EP_F32_DECAY    = 4,   // outF = exp(-exp(-softplus(-(bias[n]+s)) - 0.5))
  EP_F32_VMIX     = 5,   // outF = vraw + (v_first - vraw)*sigmoid(bias[n]+s)
  EP_F32_NONE     = 6    // outF = s
};

template <int EPI>
__device__ inline void store_elem(float s, size_t idx, int gc,
                                  const float* bias, const float* auxF,
                                  const __bf16* auxB, float* outF,
                                  __bf16* outB) {
  if (EPI == EP_BF16_NONE) {
    outB[idx] = (__bf16)s;
  } else if (EPI == EP_BF16_TANH) {
    outB[idx] = (__bf16)tanhf(s);
  } else if (EPI == EP_BF16_SIG) {
    outB[idx] = (__bf16)(1.f / (1.f + expf(-s)));
  } else if (EPI == EP_BF16_SIGBIAS) {
    outB[idx] = (__bf16)(1.f / (1.f + expf(-(s + bias[gc]))));
  } else if (EPI == EP_F32_DECAY) {
    float u = s + bias[gc];
    float w = -log1pf(expf(-u)) - 0.5f;   // -softplus(-u) - 0.5
    outF[idx] = expf(-expf(w));
  } else if (EPI == EP_F32_VMIX) {
    float u  = s + bias[gc];
    float sg = 1.f / (1.f + expf(-u));
    float vr = (float)auxB[idx];
    outF[idx] = vr + (auxF[idx] - vr) * sg;
  } else {
    outF[idx] = s;
  }
}

// ---------------------------------------------------------------------------
// GEMM 1: OUT[M,N] = mix(x)[M,K] @ W^T,  A(m,k) = x + (shift(x)-x)*mixc[k]
// W given pre-converted bf16, row-major [N,K]. Block: 128 thr, tile 64x64x32.
// Branch-free shifted load: prev-row pointer clamps to current row at s==0 and
// the loaded value is scaled by 0 there, so loads are unconditional b128s.
// ---------------------------------------------------------------------------
template <int EPI>
__global__ __launch_bounds__(128) void gemm_mix_kernel(
    const float* __restrict__ X, const float* __restrict__ mixc,
    const __bf16* __restrict__ Wt, __bf16* __restrict__ outB,
    const int N, const int K) {
  __shared__ __align__(16) __bf16 As[64 * 32];
  __shared__ __align__(16) __bf16 Bs[64 * 32];
  const int t    = threadIdx.x;
  const int lane = t & 31;
  const int wv   = t >> 5;
  const int wm   = wv >> 1, wn = wv & 1;
  const int bm   = blockIdx.y * 64;
  const int bn   = blockIdx.x * 64;
  const int kq   = (t & 3) << 3;   // 0,8,16,24
  const int r0   = t >> 2;         // 0..31

  // Per-thread row constants (rows rr=r0 and rr=r0+32)
  const int gm0 = bm + r0;
  const int gm1 = gm0 + 32;
  const bool f0 = ((gm0 & (S_LEN - 1)) == 0);
  const bool f1 = ((gm1 & (S_LEN - 1)) == 0);
  const float ps0 = f0 ? 0.f : 1.f;
  const float ps1 = f1 ? 0.f : 1.f;
  const size_t prow0 = (size_t)(f0 ? gm0 : gm0 - 1) * C_DIM;
  const size_t prow1 = (size_t)(f1 ? gm1 : gm1 - 1) * C_DIM;
  const size_t xrow0 = (size_t)gm0 * C_DIM;
  const size_t xrow1 = (size_t)gm1 * C_DIM;

  v8f acc[2][2] = {};

  for (int k0 = 0; k0 < K; k0 += 32) {
    const int kc = k0 + kq;
    // ---- A tile: fused token-shift mix, f32 -> bf16, branch-free ----
    {
      const v4f ma = *(const v4f*)(mixc + kc);
      const v4f mb = *(const v4f*)(mixc + kc + 4);
      // row r0
      {
        const v4f xa = *(const v4f*)(X + xrow0 + kc);
        const v4f xb = *(const v4f*)(X + xrow0 + kc + 4);
        const v4f pa = *(const v4f*)(X + prow0 + kc);
        const v4f pb = *(const v4f*)(X + prow0 + kc + 4);
        union { v8bf v; __bf16 e[8]; } pack;
        #pragma unroll
        for (int e = 0; e < 4; ++e) {
          pack.e[e]     = (__bf16)(xa[e] + (pa[e] * ps0 - xa[e]) * ma[e]);
          pack.e[e + 4] = (__bf16)(xb[e] + (pb[e] * ps0 - xb[e]) * mb[e]);
        }
        *(v8bf*)(&As[r0 * 32 + kq]) = pack.v;
      }
      // row r0+32
      {
        const v4f xa = *(const v4f*)(X + xrow1 + kc);
        const v4f xb = *(const v4f*)(X + xrow1 + kc + 4);
        const v4f pa = *(const v4f*)(X + prow1 + kc);
        const v4f pb = *(const v4f*)(X + prow1 + kc + 4);
        union { v8bf v; __bf16 e[8]; } pack;
        #pragma unroll
        for (int e = 0; e < 4; ++e) {
          pack.e[e]     = (__bf16)(xa[e] + (pa[e] * ps1 - xa[e]) * ma[e]);
          pack.e[e + 4] = (__bf16)(xb[e] + (pb[e] * ps1 - xb[e]) * mb[e]);
        }
        *(v8bf*)(&As[(r0 + 32) * 32 + kq]) = pack.v;
      }
    }
    // ---- B tile: bf16 weights, async -> LDS when available ----
    #pragma unroll
    for (int rr = r0; rr < 64; rr += 32) {
      const int gn = bn + rr;
      __bf16* dst = &Bs[rr * 32 + kq];
      if (gn < N) {
        const __bf16* src = Wt + (size_t)gn * K + (size_t)kc;
#if ASYNC_OK
        __builtin_amdgcn_global_load_async_to_lds_b128((AS1 v4i*)src,
                                                       (AS3 v4i*)dst, 0, 0);
#else
        *(uint4*)dst = *(const uint4*)src;
#endif
      } else {
        uint4 z = {0, 0, 0, 0};
        *(uint4*)dst = z;
      }
    }
#if ASYNC_OK
    __builtin_amdgcn_s_wait_asynccnt(0);
#endif
    __syncthreads();

    v16bf a0 = frag_ld(As, wm * 32, lane);
    v16bf a1 = frag_ld(As, wm * 32 + 16, lane);
    v16bf b0 = frag_ld(Bs, wn * 32, lane);
    v16bf b1 = frag_ld(Bs, wn * 32 + 16, lane);
    acc[0][0] = wmma_bf16(a0, b0, acc[0][0]);
    acc[0][1] = wmma_bf16(a0, b1, acc[0][1]);
    acc[1][0] = wmma_bf16(a1, b0, acc[1][0]);
    acc[1][1] = wmma_bf16(a1, b1, acc[1][1]);
    __syncthreads();
  }

  // ---- store: lane L -> col L%16, rows (L/16)*8 + p (per C/D VGPR layout) ----
  const int cn = lane & 15;
  const int rb = (lane >> 4) << 3;
  #pragma unroll
  for (int ti = 0; ti < 2; ++ti)
    #pragma unroll
    for (int tj = 0; tj < 2; ++tj) {
      const int gc = bn + wn * 32 + tj * 16 + cn;
      if (gc >= N) continue;
      #pragma unroll
      for (int p = 0; p < 8; ++p) {
        const int gr = bm + wm * 32 + ti * 16 + rb + p;
        const size_t idx = (size_t)gr * N + gc;
        store_elem<EPI>(acc[ti][tj][p], idx, gc, nullptr, nullptr, nullptr,
                        nullptr, outB);
      }
    }
}

// ---------------------------------------------------------------------------
// GEMM 2: OUT[M,N] = A[M,K](bf16) @ W^T (bf16 [N,K]) with epilogue.
// ---------------------------------------------------------------------------
template <int EPI>
__global__ __launch_bounds__(128) void gemm_plain_kernel(
    const __bf16* __restrict__ A, const __bf16* __restrict__ Wt,
    const float* __restrict__ bias, const float* __restrict__ auxF,
    const __bf16* __restrict__ auxB, float* __restrict__ outF,
    __bf16* __restrict__ outB, const int N, const int K) {
  __shared__ __align__(16) __bf16 As[64 * 32];
  __shared__ __align__(16) __bf16 Bs[64 * 32];
  const int t    = threadIdx.x;
  const int lane = t & 31;
  const int wv   = t >> 5;
  const int wm   = wv >> 1, wn = wv & 1;
  const int bm   = blockIdx.y * 64;
  const int bn   = blockIdx.x * 64;
  const int kq   = (t & 3) << 3;
  const int r0   = t >> 2;

  v8f acc[2][2] = {};

  for (int k0 = 0; k0 < K; k0 += 32) {
    #pragma unroll
    for (int rr = r0; rr < 64; rr += 32) {
      const int gm = bm + rr;   // always < M
      const __bf16* src = A + (size_t)gm * K + (size_t)(k0 + kq);
      __bf16* dst = &As[rr * 32 + kq];
#if ASYNC_OK
      __builtin_amdgcn_global_load_async_to_lds_b128((AS1 v4i*)src,
                                                     (AS3 v4i*)dst, 0, 0);
#else
      *(uint4*)dst = *(const uint4*)src;
#endif
    }
    #pragma unroll
    for (int rr = r0; rr < 64; rr += 32) {
      const int gn = bn + rr;
      __bf16* dst = &Bs[rr * 32 + kq];
      if (gn < N) {
        const __bf16* src = Wt + (size_t)gn * K + (size_t)(k0 + kq);
#if ASYNC_OK
        __builtin_amdgcn_global_load_async_to_lds_b128((AS1 v4i*)src,
                                                       (AS3 v4i*)dst, 0, 0);
#else
        *(uint4*)dst = *(const uint4*)src;
#endif
      } else {
        uint4 z = {0, 0, 0, 0};
        *(uint4*)dst = z;
      }
    }
#if ASYNC_OK
    __builtin_amdgcn_s_wait_asynccnt(0);
#endif
    __syncthreads();

    v16bf a0 = frag_ld(As, wm * 32, lane);
    v16bf a1 = frag_ld(As, wm * 32 + 16, lane);
    v16bf b0 = frag_ld(Bs, wn * 32, lane);
    v16bf b1 = frag_ld(Bs, wn * 32 + 16, lane);
    acc[0][0] = wmma_bf16(a0, b0, acc[0][0]);
    acc[0][1] = wmma_bf16(a0, b1, acc[0][1]);
    acc[1][0] = wmma_bf16(a1, b0, acc[1][0]);
    acc[1][1] = wmma_bf16(a1, b1, acc[1][1]);
    __syncthreads();
  }

  const int cn = lane & 15;
  const int rb = (lane >> 4) << 3;
  #pragma unroll
  for (int ti = 0; ti < 2; ++ti)
    #pragma unroll
    for (int tj = 0; tj < 2; ++tj) {
      const int gc = bn + wn * 32 + tj * 16 + cn;
      if (gc >= N) continue;
      #pragma unroll
      for (int p = 0; p < 8; ++p) {
        const int gr = bm + wm * 32 + ti * 16 + rb + p;
        const size_t idx = (size_t)gr * N + gc;
        store_elem<EPI>(acc[ti][tj][p], idx, gc, bias, auxF, auxB, outF, outB);
      }
    }
}

// ---------------------------------------------------------------------------
// Weight conversion kernels (f32 -> bf16, optional transpose)
// ---------------------------------------------------------------------------
__global__ void conv_bf16(const float* __restrict__ in,
                          __bf16* __restrict__ out, size_t n) {
  size_t i = (size_t)blockIdx.x * blockDim.x + threadIdx.x;
  if (i < n) out[i] = (__bf16)in[i];
}

// in [R, Cc] row-major  ->  out [Cc, R] row-major
__global__ void conv_bf16_t(const float* __restrict__ in,
                            __bf16* __restrict__ out, int R, int Cc) {
  size_t i = (size_t)blockIdx.x * blockDim.x + threadIdx.x;
  if (i < (size_t)R * Cc) {
    int r = (int)(i / Cc), c = (int)(i % Cc);
    out[(size_t)c * R + r] = (__bf16)in[i];
  }
}

// ---------------------------------------------------------------------------
// k post-processing: kk = L2normalize_perhead(k * k_k); kmod = k*(1+(a-1)*k_a)
// One wave per (b,s,h) row of 64; lane handles elems lane and lane+32.
// ---------------------------------------------------------------------------
__global__ __launch_bounds__(128) void kpost_kernel(
    const __bf16* __restrict__ Kraw, const __bf16* __restrict__ A,
    const float* __restrict__ k_k, const float* __restrict__ k_a,
    __bf16* __restrict__ KK, __bf16* __restrict__ Km) {
  const int lane = threadIdx.x & 31;
  const size_t row = (size_t)blockIdx.x * 4 + (threadIdx.x >> 5);
  const size_t m = row >> 5;          // / H
  const int h = (int)(row & 31);
  const size_t base = m * C_DIM + (size_t)h * 64;
  const int c0 = h * 64 + lane, c1 = c0 + 32;

  const float k0v = (float)Kraw[base + lane];
  const float k1v = (float)Kraw[base + 32 + lane];
  const float p0 = k0v * k_k[c0], p1 = k1v * k_k[c1];
  float ss = p0 * p0 + p1 * p1;
  #pragma unroll
  for (int o = 16; o > 0; o >>= 1) ss += __shfl_xor(ss, o, 32);
  const float invn = 1.f / fmaxf(sqrtf(ss), 1e-12f);
  KK[base + lane]      = (__bf16)(p0 * invn);
  KK[base + 32 + lane] = (__bf16)(p1 * invn);

  const float a0v = (float)A[base + lane], a1v = (float)A[base + 32 + lane];
  Km[base + lane]      = (__bf16)(k0v * (1.f + (a0v - 1.f) * k_a[c0]));
  Km[base + 32 + lane] = (__bf16)(k1v * (1.f + (a1v - 1.f) * k_a[c1]));
}

// ---------------------------------------------------------------------------
// WKV-7 recurrence. One block per (b,h); 64 threads; thread i owns state row i
// (64 f32 in VGPRs). Per-step vectors broadcast through LDS.
//   sa_i   = sum_j st[i][j] * (-kk_j)
//   st[i][j] = st[i][j]*w_j + sa_i*(kk_j*a_j) + v_i*k_j
//   y_i    = sum_j st[i][j]*r_j
// ---------------------------------------------------------------------------
__global__ __launch_bounds__(64) void wkv_kernel(
    const __bf16* __restrict__ R, const float* __restrict__ Wd,
    const __bf16* __restrict__ K, const float* __restrict__ V,
    const __bf16* __restrict__ KK, const __bf16* __restrict__ A,
    float* __restrict__ Y) {
  const int bh = blockIdx.x;     // 0..B*H-1
  const int b  = bh >> 5;
  const int h  = bh & 31;
  const int j  = threadIdx.x;    // also row index i

  __shared__ float shR[64], shW[64], shK[64], shV[64], shKK[64], shA[64];
  float st[64];
  #pragma unroll
  for (int q = 0; q < 64; ++q) st[q] = 0.f;

  size_t idx = (size_t)b * S_LEN * C_DIM + (size_t)h * 64 + j;
  for (int t = 0; t < S_LEN; ++t, idx += C_DIM) {
    shR[j]  = (float)R[idx];
    shW[j]  = Wd[idx];
    shK[j]  = (float)K[idx];
    shV[j]  = V[idx];
    shKK[j] = (float)KK[idx];
    shA[j]  = (float)A[idx];
    __syncthreads();

    float sa = 0.f;
    #pragma unroll 8
    for (int q = 0; q < 64; ++q) sa -= st[q] * shKK[q];
    const float vi = shV[j];
    float yv = 0.f;
    #pragma unroll 8
    for (int q = 0; q < 64; ++q) {
      float bb = shKK[q] * shA[q];
      float s2 = st[q] * shW[q] + sa * bb + vi * shK[q];
      st[q] = s2;
      yv += s2 * shR[q];
    }
    Y[idx] = yv;
    __syncthreads();
  }
}

// ---------------------------------------------------------------------------
// GroupNorm (per (b,s,h) over N=64) + bonus + gate:
//   z = ((y-mu)/sqrt(var+eps)*ln_w + ln_b + (sum r*k*r_k)*v) * g   -> bf16
// ---------------------------------------------------------------------------
__global__ __launch_bounds__(128) void gn_kernel(
    const float* __restrict__ Y, const __bf16* __restrict__ R,
    const __bf16* __restrict__ Km, const float* __restrict__ V,
    const __bf16* __restrict__ G, const float* __restrict__ r_k,
    const float* __restrict__ lnw, const float* __restrict__ lnb,
    __bf16* __restrict__ Z) {
  const int lane = threadIdx.x & 31;
  const size_t row = (size_t)blockIdx.x * 4 + (threadIdx.x >> 5);
  const size_t m = row >> 5;
  const int h = (int)(row & 31);
  const size_t base = m * C_DIM + (size_t)h * 64;
  const int c0 = h * 64 + lane, c1 = c0 + 32;

  const float y0 = Y[base + lane], y1 = Y[base + 32 + lane];
  float s = y0 + y1;
  #pragma unroll
  for (int o = 16; o > 0; o >>= 1) s += __shfl_xor(s, o, 32);
  const float mean = s * (1.f / 64.f);
  const float d0 = y0 - mean, d1 = y1 - mean;
  float vs = d0 * d0 + d1 * d1;
  #pragma unroll
  for (int o = 16; o > 0; o >>= 1) vs += __shfl_xor(vs, o, 32);
  const float inv = rsqrtf(vs * (1.f / 64.f) + 0.00064f);
  float bon = (float)R[base + lane] * (float)Km[base + lane] * r_k[c0] +
              (float)R[base + 32 + lane] * (float)Km[base + 32 + lane] * r_k[c1];
  #pragma unroll
  for (int o = 16; o > 0; o >>= 1) bon += __shfl_xor(bon, o, 32);

  float z0 = (d0 * inv * lnw[c0] + lnb[c0] + bon * V[base + lane]) *
             (float)G[base + lane];
  float z1 = (d1 * inv * lnw[c1] + lnb[c1] + bon * V[base + 32 + lane]) *
             (float)G[base + 32 + lane];
  Z[base + lane]      = (__bf16)z0;
  Z[base + 32 + lane] = (__bf16)z1;
}

// ---------------------------------------------------------------------------
// Host launcher
// ---------------------------------------------------------------------------
extern "C" void kernel_launch(void* const* d_in, const int* in_sizes, int n_in,
                              void* d_out, int out_size, void* d_ws,
                              size_t ws_size, hipStream_t stream) {
  const float* x      = (const float*)d_in[0];
  const float* vfirst = (const float*)d_in[1];
  const float* x_r    = (const float*)d_in[2];
  const float* x_w    = (const float*)d_in[3];
  const float* x_k    = (const float*)d_in[4];
  const float* x_v    = (const float*)d_in[5];
  const float* x_a    = (const float*)d_in[6];
  const float* x_g    = (const float*)d_in[7];
  const float* w0     = (const float*)d_in[8];
  const float* w1     = (const float*)d_in[9];
  const float* w2     = (const float*)d_in[10];
  const float* a0     = (const float*)d_in[11];
  const float* a1     = (const float*)d_in[12];
  const float* a2     = (const float*)d_in[13];
  const float* v0     = (const float*)d_in[14];
  const float* v1     = (const float*)d_in[15];
  const float* v2     = (const float*)d_in[16];
  const float* g1     = (const float*)d_in[17];
  const float* g2     = (const float*)d_in[18];
  const float* k_k    = (const float*)d_in[19];
  const float* k_a    = (const float*)d_in[20];
  const float* r_k    = (const float*)d_in[21];
  const float* Wr     = (const float*)d_in[22];
  const float* Wk     = (const float*)d_in[23];
  const float* Wv     = (const float*)d_in[24];
  const float* Wo     = (const float*)d_in[25];
  const float* ln_w   = (const float*)d_in[26];
  const float* ln_b   = (const float*)d_in[27];

  uint8_t* base = (uint8_t*)d_ws;
  size_t off = 0;
  auto alloc = [&](size_t bytes) -> void* {
    void* p = base + off;
    off += (bytes + 255) & ~(size_t)255;
    return p;
  };

  const size_t MC = (size_t)Mrows * C_DIM;
  // bf16 weights
  __bf16* wr_bf = (__bf16*)alloc((size_t)C_DIM * C_DIM * 2);
  __bf16* wk_bf = (__bf16*)alloc((size_t)C_DIM * C_DIM * 2);
  __bf16* wv_bf = (__bf16*)alloc((size_t)C_DIM * C_DIM * 2);
  __bf16* wo_bf = (__bf16*)alloc((size_t)C_DIM * C_DIM * 2);
  __bf16* w1T = (__bf16*)alloc((size_t)DWd * C_DIM * 2);   // [DW,C]
  __bf16* w2T = (__bf16*)alloc((size_t)C_DIM * DWd * 2);   // [C,DW]
  __bf16* a1T = (__bf16*)alloc((size_t)DAd * C_DIM * 2);
  __bf16* a2T = (__bf16*)alloc((size_t)C_DIM * DAd * 2);
  __bf16* v1T = (__bf16*)alloc((size_t)DVd * C_DIM * 2);
  __bf16* v2T = (__bf16*)alloc((size_t)C_DIM * DVd * 2);
  __bf16* g1T = (__bf16*)alloc((size_t)DGd * C_DIM * 2);
  __bf16* g2T = (__bf16*)alloc((size_t)C_DIM * DGd * 2);
  // activations
  __bf16* r_bf    = (__bf16*)alloc(MC * 2);
  __bf16* kraw_bf = (__bf16*)alloc(MC * 2);
  __bf16* vraw_bf = (__bf16*)alloc(MC * 2);
  __bf16* a_bf    = (__bf16*)alloc(MC * 2);
  __bf16* g_bf    = (__bf16*)alloc(MC * 2);
  __bf16* kk_bf   = (__bf16*)alloc(MC * 2);
  __bf16* kmod_bf = (__bf16*)alloc(MC * 2);
  __bf16* z_bf    = (__bf16*)alloc(MC * 2);
  __bf16* hw_bf   = (__bf16*)alloc((size_t)Mrows * DWd * 2);
  __bf16* ha_bf   = (__bf16*)alloc((size_t)Mrows * DAd * 2);
  __bf16* hv_bf   = (__bf16*)alloc((size_t)Mrows * DVd * 2);
  __bf16* hg_bf   = (__bf16*)alloc((size_t)Mrows * DGd * 2);
  float* decay_f  = (float*)alloc(MC * 4);
  float* v_f      = (float*)alloc(MC * 4);
  float* y_f      = (float*)alloc(MC * 4);

  auto cg = [](size_t n) { return dim3((unsigned)((n + 255) / 256)); };

  // 1) weight conversions
  conv_bf16<<<cg((size_t)C_DIM * C_DIM), 256, 0, stream>>>(Wr, wr_bf, (size_t)C_DIM * C_DIM);
  conv_bf16<<<cg((size_t)C_DIM * C_DIM), 256, 0, stream>>>(Wk, wk_bf, (size_t)C_DIM * C_DIM);
  conv_bf16<<<cg((size_t)C_DIM * C_DIM), 256, 0, stream>>>(Wv, wv_bf, (size_t)C_DIM * C_DIM);
  conv_bf16<<<cg((size_t)C_DIM * C_DIM), 256, 0, stream>>>(Wo, wo_bf, (size_t)C_DIM * C_DIM);
  conv_bf16_t<<<cg((size_t)C_DIM * DWd), 256, 0, stream>>>(w1, w1T, C_DIM, DWd);
  conv_bf16_t<<<cg((size_t)C_DIM * DWd), 256, 0, stream>>>(w2, w2T, DWd, C_DIM);
  conv_bf16_t<<<cg((size_t)C_DIM * DAd), 256, 0, stream>>>(a1, a1T, C_DIM, DAd);
  conv_bf16_t<<<cg((size_t)C_DIM * DAd), 256, 0, stream>>>(a2, a2T, DAd, C_DIM);
  conv_bf16_t<<<cg((size_t)C_DIM * DVd), 256, 0, stream>>>(v1, v1T, C_DIM, DVd);
  conv_bf16_t<<<cg((size_t)C_DIM * DVd), 256, 0, stream>>>(v2, v2T, DVd, C_DIM);
  conv_bf16_t<<<cg((size_t)C_DIM * DGd), 256, 0, stream>>>(g1, g1T, C_DIM, DGd);
  conv_bf16_t<<<cg((size_t)C_DIM * DGd), 256, 0, stream>>>(g2, g2T, DGd, C_DIM);

  // 2) mix-fused WMMA GEMMs
  dim3 blk(128);
  dim3 gC((C_DIM + 63) / 64, Mrows / 64);
  dim3 gW((DWd + 63) / 64, Mrows / 64);
  dim3 gA((DAd + 63) / 64, Mrows / 64);
  dim3 gV((DVd + 63) / 64, Mrows / 64);
  dim3 gG((DGd + 63) / 64, Mrows / 64);
  gemm_mix_kernel<EP_BF16_NONE><<<gC, blk, 0, stream>>>(x, x_r, wr_bf, r_bf, C_DIM, C_DIM);
  gemm_mix_kernel<EP_BF16_NONE><<<gC, blk, 0, stream>>>(x, x_k, wk_bf, kraw_bf, C_DIM, C_DIM);
  gemm_mix_kernel<EP_BF16_NONE><<<gC, blk, 0, stream>>>(x, x_v, wv_bf, vraw_bf, C_DIM, C_DIM);
  gemm_mix_kernel<EP_BF16_TANH><<<gW, blk, 0, stream>>>(x, x_w, w1T, hw_bf, DWd, C_DIM);
  gemm_mix_kernel<EP_BF16_NONE><<<gA, blk, 0, stream>>>(x, x_a, a1T, ha_bf, DAd, C_DIM);
  gemm_mix_kernel<EP_BF16_NONE><<<gV, blk, 0, stream>>>(x, x_v, v1T, hv_bf, DVd, C_DIM);
  gemm_mix_kernel<EP_BF16_SIG><<<gG, blk, 0, stream>>>(x, x_g, g1T, hg_bf, DGd, C_DIM);

  // 3) LoRA second GEMMs with fused epilogues
  gemm_plain_kernel<EP_F32_DECAY><<<gC, blk, 0, stream>>>(
      hw_bf, w2T, w0, nullptr, nullptr, decay_f, nullptr, C_DIM, DWd);
  gemm_plain_kernel<EP_BF16_SIGBIAS><<<gC, blk, 0, stream>>>(
      ha_bf, a2T, a0, nullptr, nullptr, nullptr, a_bf, C_DIM, DAd);
  gemm_plain_kernel<EP_F32_VMIX><<<gC, blk, 0, stream>>>(
      hv_bf, v2T, v0, vfirst, vraw_bf, v_f, nullptr, C_DIM, DVd);
  gemm_plain_kernel<EP_BF16_NONE><<<gC, blk, 0, stream>>>(
      hg_bf, g2T, nullptr, nullptr, nullptr, nullptr, g_bf, C_DIM, DGd);

  // 4) k post (kk normalize + k_a modulation)
  kpost_kernel<<<(Mrows * Hh) / 4, 128, 0, stream>>>(kraw_bf, a_bf, k_k, k_a,
                                                     kk_bf, kmod_bf);
  // 5) sequential WKV recurrence
  wkv_kernel<<<Bsz * Hh, 64, 0, stream>>>(r_bf, decay_f, kmod_bf, v_f, kk_bf,
                                          a_bf, y_f);
  // 6) groupnorm + bonus + gate
  gn_kernel<<<(Mrows * Hh) / 4, 128, 0, stream>>>(y_f, r_bf, kmod_bf, v_f,
                                                  g_bf, r_k, ln_w, ln_b, z_bf);
  // 7) output projection -> d_out (f32)
  gemm_plain_kernel<EP_F32_NONE><<<gC, blk, 0, stream>>>(
      z_bf, wo_bf, nullptr, nullptr, nullptr, (float*)d_out, nullptr, C_DIM,
      C_DIM);
  // 8) second tuple output: v_first passthrough
  (void)hipMemcpyAsync((float*)d_out + MC, vfirst, MC * sizeof(float),
                       hipMemcpyDeviceToDevice, stream);
  (void)in_sizes; (void)n_in; (void)out_size; (void)ws_size;
}
